// AttentionBlock_18631568130948
// MI455X (gfx1250) — matrix operand order repaired and checked by reading, not compiled, not run on hardware
//
#include <hip/hip_runtime.h>
#include <hip/hip_bf16.h>

// ---------------------------------------------------------------------------
// AttentionBlock for MI455X (gfx1250, wave32, WMMA).
// B=8, C=512, L=1024, H=8 heads, ch=64, G=32 groups.
// All GEMMs via v_wmma_f32_16x16x32_f16 (f16 in, f32 accumulate).
// B-operand tiles are staged in LDS in fragment-ready order so every WMMA
// operand is loaded with b128 vector ops (no scalar gathers).
// ---------------------------------------------------------------------------

typedef __attribute__((ext_vector_type(16))) _Float16 v16h;
typedef __attribute__((ext_vector_type(8)))  float    v8f;
typedef __attribute__((ext_vector_type(4)))  int      v4i;

#define WMMA_F16(A, B, Cacc) \
  __builtin_amdgcn_wmma_f32_16x16x32_f16(false, (A), false, (B), (short)0, (Cacc), false, false)

// 16-half fragment viewed as two 16B chunks for vectorized b128 loads.
union Frag16 {
  v16h h;
  uint4 u[2];
};

// Load a fragment stored contiguously per lane (frag-order LDS buffers, V tiles).
__device__ __forceinline__ v16h load_frag16(const _Float16* p) {
  Frag16 f;
  f.u[0] = *reinterpret_cast<const uint4*>(p);
  f.u[1] = *reinterpret_cast<const uint4*>(p + 8);
  return f.h;
}

// ---------------------------------------------------------------------------
// CDNA5 async global->LDS staging (ASYNCcnt path), with plain-copy fallback.
// Builtin signature (probe-confirmed): (v4i AS1*, v4i AS3*, Ii, Ii).
// ---------------------------------------------------------------------------
#if __has_builtin(__builtin_amdgcn_global_load_async_to_lds_b128)
#define HAVE_ASYNC_LDS 1
typedef v4i __attribute__((address_space(1))) v4i_g;
typedef v4i __attribute__((address_space(3))) v4i_l;
__device__ __forceinline__ void stage16_async(const _Float16* g, _Float16* l) {
  __builtin_amdgcn_global_load_async_to_lds_b128(
      (v4i_g*)(unsigned long long)g,
      (v4i_l*)(unsigned long long)l,
      0, 0);
}
__device__ __forceinline__ void wait_async0() {
#if __has_builtin(__builtin_amdgcn_s_wait_asynccnt)
  __builtin_amdgcn_s_wait_asynccnt(0);
#else
  asm volatile("s_wait_asynccnt 0x0" ::: "memory");
#endif
}
#else
#define HAVE_ASYNC_LDS 0
__device__ __forceinline__ void stage16_async(const _Float16* g, _Float16* l) {
  *reinterpret_cast<uint4*>(l) = *reinterpret_cast<const uint4*>(g);
}
__device__ __forceinline__ void wait_async0() {}
#endif

// ---------------------------------------------------------------------------
// Kernel 1: f32 -> f16 conversion (weights)
// ---------------------------------------------------------------------------
__global__ void f32_to_f16_kernel(const float* __restrict__ in,
                                  _Float16* __restrict__ out, int n) {
  int i = blockIdx.x * blockDim.x + threadIdx.x;
  if (i < n) out[i] = (_Float16)in[i];
}

// ---------------------------------------------------------------------------
// Kernel 2: GroupNorm. One workgroup per (batch, group); each group is a
// contiguous 16*1024 f32 slab. Writes f16 normalized activations.
// ---------------------------------------------------------------------------
__global__ void __launch_bounds__(256)
groupnorm_kernel(const float* __restrict__ x, const float* __restrict__ gw,
                 const float* __restrict__ gb, _Float16* __restrict__ xn) {
  __shared__ float red[256];
  __shared__ float red2[256];
  const int b = blockIdx.x >> 5;   // batch 0..7
  const int g = blockIdx.x & 31;   // group 0..31
  const long base = ((long)b * 512 + (long)g * 16) * 1024;

  float s = 0.f, s2 = 0.f;
  for (int i = threadIdx.x; i < 16384; i += 256) {
    float v = x[base + i];
    s += v; s2 += v * v;
  }
  red[threadIdx.x] = s; red2[threadIdx.x] = s2;
  __syncthreads();
  for (int off = 128; off > 0; off >>= 1) {
    if (threadIdx.x < off) {
      red[threadIdx.x]  += red[threadIdx.x + off];
      red2[threadIdx.x] += red2[threadIdx.x + off];
    }
    __syncthreads();
  }
  const float mean = red[0] * (1.f / 16384.f);
  const float var  = red2[0] * (1.f / 16384.f) - mean * mean;
  const float rinv = rsqrtf(var + 1e-5f);

  for (int i = threadIdx.x; i < 16384; i += 256) {
    int ch = g * 16 + (i >> 10);
    float v = (x[base + i] - mean) * rinv * gw[ch] + gb[ch];
    xn[base + i] = (_Float16)v;
  }
}

// ---------------------------------------------------------------------------
// Kernel 3/5: WMMA GEMM  out[b,m,n] = sum_k W[m,k] * X[b,k,n] + bias[m]
// Block = 128 threads (4 waves). Workgroup tile: 64(M) x 32(N).
// X tile is staged in fragment-ready LDS order:
//   element (k,n) -> Xf[n>>4][ ((n&15) + 16*(k>=16)) * 16 + (k&15) ]
// so each wave reads its two B fragments as 2x b128 each. All 4 waves share
// the same B fragments; A fragments are two contiguous b128 global chunks.
// mode 0: f16 output (QKV).  mode 1: f32 output = residual + bias + result.
// ---------------------------------------------------------------------------
__global__ void __launch_bounds__(128)
wmma_gemm_kernel(const _Float16* __restrict__ W, const _Float16* __restrict__ X,
                 const float* __restrict__ bias,
                 _Float16* __restrict__ out16, float* __restrict__ out32,
                 const float* __restrict__ residual,
                 int M, int K, int N, long strideX, long strideO, int mode) {
  __shared__ __align__(16) _Float16 Xf[2][512];  // [n-half][lane*16 + elem]

  const int lane = threadIdx.x & 31;
  const int wave = threadIdx.x >> 5;
  const int hw   = lane >> 4;     // half-wave 0/1
  const int l16  = lane & 15;
  const int b    = blockIdx.z;
  const int m0   = blockIdx.y * 64 + wave * 16;
  const int n0   = blockIdx.x * 32;
  const _Float16* Xb = X + (long)b * strideX;

  const int srow  = threadIdx.x >> 2;   // staged k row 0..31
  const int spart = threadIdx.x & 3;    // 0..3 (8 halfs each)
  const int snh   = spart >> 1;                         // n-half of this chunk
  const int slane = (spart & 1) * 8 + (srow & 16);      // first dest lane
  const int selem = srow & 15;                          // dest elem (k&15)

  v8f acc0 = {}; v8f acc1 = {};

  for (int k0 = 0; k0 < K; k0 += 32) {
    // Stage X tile [32 K x 32 N] into fragment order: coalesced b128 read,
    // 8 b16 scatter stores per thread.
    const _Float16* gsrc = Xb + (long)(k0 + srow) * N + n0 + spart * 8;
    union { uint4 u; _Float16 h[8]; } ld;
    ld.u = *reinterpret_cast<const uint4*>(gsrc);
    __builtin_prefetch(gsrc + 32L * N, 0, 1);  // warm next K tile in L2
#pragma unroll
    for (int i = 0; i < 8; ++i)
      Xf[snh][(slane + i) * 16 + selem] = ld.h[i];
    __syncthreads();

    // A fragment: two contiguous 16B chunks of the weight row.
    // elems 0..15 -> K = k0 + hw*16 + e.
    Frag16 a;
    const _Float16* Wr = W + (long)(m0 + l16) * K + k0 + hw * 16;
    a.u[0] = *reinterpret_cast<const uint4*>(Wr);
    a.u[1] = *reinterpret_cast<const uint4*>(Wr + 8);

    // B fragments: contiguous per lane in frag-order LDS.
    v16h b0 = load_frag16(&Xf[0][lane * 16]);
    v16h b1 = load_frag16(&Xf[1][lane * 16]);

    acc0 = WMMA_F16(a.h, b0, acc0);
    acc1 = WMMA_F16(a.h, b1, acc1);
    __syncthreads();
  }

  // Epilogue. D layout: elem j -> row j + 8*hw, col l16.
#pragma unroll
  for (int j = 0; j < 8; ++j) {
    int m = m0 + j + hw * 8;
    int n = n0 + l16;
    float bv = bias[m];
    if (mode == 0) {
      _Float16* O = out16 + (long)b * strideO;
      O[(long)m * N + n]      = (_Float16)(acc0[j] + bv);
      O[(long)m * N + n + 16] = (_Float16)(acc1[j] + bv);
    } else {
      float* O = out32 + (long)b * strideO;
      const float* R = residual + (long)b * strideO;
      long i0 = (long)m * N + n;
      O[i0]      = R[i0]      + acc0[j] + bv;
      O[i0 + 16] = R[i0 + 16] + acc1[j] + bv;
    }
  }
}

// ---------------------------------------------------------------------------
// Kernel 4: flash attention. grid = (L/64, B*H), block = 128 (4 waves).
// Each wave: 16 query rows. Per s-step the workgroup stages:
//   K tile -> fragment-ready LDS order (4 fragments, shared by all waves)
//   V tile -> natural [c][s] layout via async global->LDS b128 (fragments
//             are already contiguous per lane in that layout).
// Online softmax in f32; P round-trips LDS for the C->A relayout.
// Folds the double sqrt(sqrt(ch)) scaling as 1/8 on scores.
// ---------------------------------------------------------------------------
__global__ void __launch_bounds__(128)
attention_kernel(const _Float16* __restrict__ qkv, _Float16* __restrict__ hout) {
  __shared__ __align__(16) _Float16 Kf[4][512];   // [cc*2+nh][lane*16 + elem]
  __shared__ __align__(16) _Float16 Vt[64 * 32];  // natural [c][s]
  __shared__ __align__(16) _Float16 Pb[4][16 * 32];

  const int lane  = threadIdx.x & 31;
  const int wave  = threadIdx.x >> 5;
  const int hw    = lane >> 4;
  const int l16   = lane & 15;
  const int bh    = blockIdx.y;       // 0..63
  const int batch = bh >> 3;
  const int head  = bh & 7;
  const int t0    = blockIdx.x * 64 + wave * 16;
  const long L    = 1024;

  const _Float16* qb = qkv + (long)batch * 1536 * L + (long)(head * 192) * L;
  const _Float16* kb = qb + 64 * L;
  const _Float16* vb = qb + 128 * L;

  // Q as two A fragments (channel chunks 0-31, 32-63); A[m=t][k=c] = q[c][t0+m].
  // Gather over channels (stride L) — done once per kernel.
  v16h aq0, aq1;
#pragma unroll
  for (int e = 0; e < 16; ++e) {
    int k = e + hw * 16;
    aq0[e] = qb[(long)k * L + t0 + l16];
    aq1[e] = qb[(long)(k + 32) * L + t0 + l16];
  }

  v8f o0 = {}, o1 = {}, o2 = {}, o3 = {};
  float mrow[8], lrow[8];
#pragma unroll
  for (int j = 0; j < 8; ++j) { mrow[j] = -1e30f; lrow[j] = 0.f; }

  for (int s0 = 0; s0 < 1024; s0 += 32) {
    // Stage K (frag order) and V (natural, async) [64 c x 32 s] tiles.
    for (int idx = threadIdx.x; idx < 256; idx += 128) {
      int c = idx >> 2, part = idx & 3;
      const _Float16* kg = kb + (long)c * L + s0 + part * 8;
      const _Float16* vg = vb + (long)c * L + s0 + part * 8;
      stage16_async(vg, &Vt[c * 32 + part * 8]);
      union { uint4 u; _Float16 h[8]; } ld;
      ld.u = *reinterpret_cast<const uint4*>(kg);
      __builtin_prefetch(kg + 32, 0, 1);  // next s tile
      __builtin_prefetch(vg + 32, 0, 1);
      int frag  = (c >> 5) * 2 + (part >> 1);        // cc*2 + nh
      int lbase = (part & 1) * 8 + ((c >> 4) & 1) * 16;
      int elem  = c & 15;
#pragma unroll
      for (int i = 0; i < 8; ++i)
        Kf[frag][(lbase + i) * 16 + elem] = ld.h[i];
    }
    wait_async0();
    __syncthreads();

    // Scores S[16 t x 32 s] = Q^T K over c=0..63; all operands b128 loads.
    v8f d0 = {}, d1 = {};
    d0 = WMMA_F16(aq0, load_frag16(&Kf[0][lane * 16]), d0);  // cc=0, nh=0
    d1 = WMMA_F16(aq0, load_frag16(&Kf[1][lane * 16]), d1);  // cc=0, nh=1
    d0 = WMMA_F16(aq1, load_frag16(&Kf[2][lane * 16]), d0);  // cc=1, nh=0
    d1 = WMMA_F16(aq1, load_frag16(&Kf[3][lane * 16]), d1);  // cc=1, nh=1

    // Online softmax per row; rows live in C-layout elem j, N across 16 lanes.
#pragma unroll
    for (int j = 0; j < 8; ++j) {
      float s0v = d0[j] * 0.125f;   // scale^2 = 1/sqrt(ch) = 1/8
      float s1v = d1[j] * 0.125f;
      float mx = fmaxf(s0v, s1v);
      mx = fmaxf(mx, __shfl_xor(mx, 1));
      mx = fmaxf(mx, __shfl_xor(mx, 2));
      mx = fmaxf(mx, __shfl_xor(mx, 4));
      mx = fmaxf(mx, __shfl_xor(mx, 8));
      float mnew = fmaxf(mrow[j], mx);
      float corr = __expf(mrow[j] - mnew);
      mrow[j] = mnew;
      float p0 = __expf(s0v - mnew);
      float p1 = __expf(s1v - mnew);
      float rs = p0 + p1;
      rs += __shfl_xor(rs, 1);
      rs += __shfl_xor(rs, 2);
      rs += __shfl_xor(rs, 4);
      rs += __shfl_xor(rs, 8);
      lrow[j] = lrow[j] * corr + rs;
      o0[j] *= corr; o1[j] *= corr; o2[j] *= corr; o3[j] *= corr;
      // Spill P to per-wave LDS in [row][col] form for C->A relayout.
      int r = j + hw * 8;
      Pb[wave][r * 32 + l16]      = (_Float16)p0;
      Pb[wave][r * 32 + 16 + l16] = (_Float16)p1;
    }
    asm volatile("s_wait_dscnt 0" ::: "memory");  // same-wave LDS RAW

    // P as A fragment: A-layout chunks at +0 and +8 -> two b128 LDS loads.
    Frag16 ap;
    {
      const _Float16* pp = &Pb[wave][l16 * 32 + hw * 16];
      ap.u[0] = *reinterpret_cast<const uint4*>(pp);
      ap.u[1] = *reinterpret_cast<const uint4*>(pp + 8);
    }

    // O[t][c] += P * V^T : B[k=s_local][n=c] = Vt[c][s_local];
    // per lane contiguous -> two b128 LDS loads per fragment.
    o0 = WMMA_F16(ap.h, load_frag16(&Vt[(0 * 16 + l16) * 32 + hw * 16]), o0);
    o1 = WMMA_F16(ap.h, load_frag16(&Vt[(1 * 16 + l16) * 32 + hw * 16]), o1);
    o2 = WMMA_F16(ap.h, load_frag16(&Vt[(2 * 16 + l16) * 32 + hw * 16]), o2);
    o3 = WMMA_F16(ap.h, load_frag16(&Vt[(3 * 16 + l16) * 32 + hw * 16]), o3);
    __syncthreads();
  }

  // Finalize: h[b, head*64 + c, t] = O[t][c] / l[t], stored f16 for proj GEMM.
#pragma unroll
  for (int j = 0; j < 8; ++j) {
    float inv = 1.f / lrow[j];
    int pos = t0 + j + hw * 8;
    long base = ((long)batch * 512 + head * 64) * 1024 + pos;
    hout[base + (long)(0 * 16 + l16) * 1024] = (_Float16)(o0[j] * inv);
    hout[base + (long)(1 * 16 + l16) * 1024] = (_Float16)(o1[j] * inv);
    hout[base + (long)(2 * 16 + l16) * 1024] = (_Float16)(o2[j] * inv);
    hout[base + (long)(3 * 16 + l16) * 1024] = (_Float16)(o3[j] * inv);
  }
}

// ---------------------------------------------------------------------------
// Launch
// ---------------------------------------------------------------------------
extern "C" void kernel_launch(void* const* d_in, const int* in_sizes, int n_in,
                              void* d_out, int out_size, void* d_ws, size_t ws_size,
                              hipStream_t stream) {
  (void)in_sizes; (void)n_in; (void)out_size; (void)ws_size;

  const float* x      = (const float*)d_in[0];   // [8,512,32,32] == [8,512,1024]
  const float* norm_w = (const float*)d_in[1];
  const float* norm_b = (const float*)d_in[2];
  const float* qkv_w  = (const float*)d_in[3];   // [1536,512]
  const float* qkv_b  = (const float*)d_in[4];   // [1536]
  const float* proj_w = (const float*)d_in[5];   // [512,512]
  const float* proj_b = (const float*)d_in[6];   // [512]
  float* out = (float*)d_out;

  // Workspace layout (bytes), all offsets 16B-aligned:
  //   xn    f16 [8][512][1024]   8,388,608
  //   qkvw  f16 [1536][512]      1,572,864
  //   projw f16 [512][512]         524,288
  //   qkvo  f16 [8][1536][1024] 25,165,824
  //   hbuf  f16 [8][512][1024]   8,388,608   (total ~42 MB)
  char* ws = (char*)d_ws;
  _Float16* xn    = (_Float16*)(ws);
  _Float16* qkvw  = (_Float16*)(ws + 8388608UL);
  _Float16* projw = (_Float16*)(ws + 8388608UL + 1572864UL);
  _Float16* qkvo  = (_Float16*)(ws + 8388608UL + 1572864UL + 524288UL);
  _Float16* hbuf  = (_Float16*)(ws + 8388608UL + 1572864UL + 524288UL + 25165824UL);

  f32_to_f16_kernel<<<(786432 + 255) / 256, 256, 0, stream>>>(qkv_w, qkvw, 786432);
  f32_to_f16_kernel<<<(262144 + 255) / 256, 256, 0, stream>>>(proj_w, projw, 262144);

  groupnorm_kernel<<<256, 256, 0, stream>>>(x, norm_w, norm_b, xn);

  // QKV: M=1536, K=512, N=1024, 8 batches.
  wmma_gemm_kernel<<<dim3(32, 24, 8), 128, 0, stream>>>(
      qkvw, xn, qkv_b, qkvo, nullptr, nullptr,
      1536, 512, 1024, 512L * 1024, 1536L * 1024, 0);

  attention_kernel<<<dim3(16, 64), 128, 0, stream>>>(qkvo, hbuf);

  // Proj + bias + residual: M=512, K=512, N=1024, 8 batches, f32 out.
  wmma_gemm_kernel<<<dim3(32, 8, 8), 128, 0, stream>>>(
      projw, hbuf, proj_b, nullptr, out, x,
      512, 512, 1024, 512L * 1024, 512L * 1024, 1);
}
